// SpocoContrastiveLoss_69234872812297
// MI455X (gfx1250) — matrix operand order repaired and checked by reading, not compiled, not run on hardware
//
#include <hip/hip_runtime.h>
#include <math.h>

// CDNA5 WMMA vector types (probe-confirmed signatures)
typedef __attribute__((ext_vector_type(16))) _Float16 v16h;
typedef __attribute__((ext_vector_type(8)))  float    v8f;
// TDM descriptor vector types (6-arg builtin on this toolchain)
typedef __attribute__((ext_vector_type(4))) unsigned u32x4;
typedef __attribute__((ext_vector_type(8))) int      i32x8;
typedef __attribute__((ext_vector_type(4))) int      i32x4;

#define NITEMS 4
#define EDIM   64
#define CDIM   64
#define PPIX   (512*512)
#define TILE   128
#define ITERS  8          // tiles per block
#define NBLK   256        // blocks per pass per item (256*8*128 = 262144 = PPIX)

// workspace layout (floats) per item
#define ITEM_STRIDE 11584
#define OFF_CNT  0        // [64]   counts
#define OFF_SUM  64       // [4096] segment sums
#define OFF_MEAN 4160     // [4096] f32 means
#define OFF_M2   8256     // [64]   ||mean||^2
#define OFF_A    8320     // [3072 u32] A fragments: ks0/ks1 = -2*means (f16 pairs), ks2 = [m2 | 1]
#define OFF_ACC  11392    // [0]=var, [1]=dist, [2]=reg, [3..66]=intersect, [67..130]=sum pmap^2

// 1/two_sigma with two_sigma = 0.25/(-ln 0.9)
#define INV2SIG 0.42144206f

// ---- Tensor Data Mover: DMA one (EDIM x TILE) f32 tile global->LDS ----
// 2D descriptor per ISA ch.8: tile rows are TILE contiguous f32 pixels,
// row stride = PPIX elements. Issued by one wave; tracked by TENSORcnt.
__device__ __forceinline__ void tdm_load_tile(const float* gsrc, float* lds_dst) {
  const unsigned long long ga = (unsigned long long)(uintptr_t)gsrc;
  const unsigned lds = (unsigned)(uintptr_t)lds_dst;   // low 32 bits = LDS byte offset
  u32x4 g0;
  g0[0] = 1u;                                          // count=1 (valid user D#)
  g0[1] = lds;                                         // lds_addr
  g0[2] = (unsigned)(ga & 0xFFFFFFFFu);                // global_addr[31:0]
  g0[3] = (unsigned)((ga >> 32) & 0x01FFFFFFu) | (2u << 30);  // global_addr[56:32] | type=2
  i32x8 g1;
  g1[0] = (int)(2u << 16);                             // data_size=4B, no multicast
  g1[1] = (int)((PPIX & 0xFFFFu) << 16);               // tensor_dim0[15:0]
  g1[2] = (int)(((unsigned)PPIX >> 16) | ((unsigned)EDIM << 16)); // tensor_dim0[31:16] | tensor_dim1[15:0]
  g1[3] = (int)((unsigned)TILE << 16);                 // tensor_dim1[31:16]=0 | tile_dim0=TILE
  g1[4] = (int)EDIM;                                   // tile_dim1=EDIM, tile_dim2=0 (2D)
  g1[5] = (int)PPIX;                                   // tensor_dim0_stride[31:0]
  g1[6] = 0;                                           // stride msbs / dim1_stride (unused, 2D)
  g1[7] = 0;
  i32x4 z4 = { 0, 0, 0, 0 };
  i32x8 z8 = { 0, 0, 0, 0, 0, 0, 0, 0 };
  __builtin_amdgcn_tensor_load_to_lds(g0, g1, z4, z4, z8, 0);
}

__global__ void zero_ws_kernel(float* __restrict__ ws, int n) {
  for (int i = blockIdx.x * blockDim.x + threadIdx.x; i < n; i += gridDim.x * blockDim.x)
    ws[i] = 0.0f;
}

// ---- pass 1: segment sums via WMMA: sums(CxE) = onehot(CxP) @ e^T(PxE) ----
// TDM double-buffers the e-tiles; f32 accumulators persist across all tiles.
__global__ void stats_kernel(const float* __restrict__ emb, const int* __restrict__ tgt,
                             float* __restrict__ ws, int n) {
  __shared__ __align__(16) float etile[2][EDIM * TILE];   // 2 x 32 KB double buffer
  __shared__ int   scls[TILE];
  __shared__ float lcnt[CDIM];
  const int tid  = threadIdx.x;
  const int lane = tid & 31;
  const int g    = lane >> 4;
  const int wv   = tid >> 5;            // 8 waves
  const int mt   = wv >> 1;             // class tile 0..3
  const int dh   = wv & 1;              // dim half 0..1
  const int m    = mt * 16 + (lane & 15);
  const int kb   = g * 8;
  if (tid < CDIM) lcnt[tid] = 0.0f;
  const size_t ebase = (size_t)n * EDIM * PPIX;
  const int tile0 = blockIdx.x * ITERS;
  if (wv == 0) tdm_load_tile(emb + ebase + (size_t)tile0 * TILE, &etile[0][0]);
  v8f acc0 = {}, acc1 = {};
  for (int it = 0; it < ITERS; ++it) {
    const int p0 = (tile0 + it) * TILE;
    if (wv == 0) __builtin_amdgcn_s_wait_tensorcnt(0);   // current buffer landed
    __syncthreads();                                     // publish; prev reads done
    if (wv == 0 && it + 1 < ITERS)                       // overlap next DMA with compute
      tdm_load_tile(emb + ebase + (size_t)(p0 + TILE), &etile[(it + 1) & 1][0]);
    if (tid < TILE) scls[tid] = tgt[n * PPIX + p0 + tid];
    __syncthreads();
    if (tid < TILE) atomicAdd(&lcnt[scls[tid]], 1.0f);
    const float* et = &etile[it & 1][0];
    #pragma unroll
    for (int ks = 0; ks < 4; ++ks) {                     // K = 128 pixels in 4 steps
      union { v16h h; unsigned u[8]; } af;               // one-hot A (integer selects)
      #pragma unroll
      for (int v = 0; v < 8; ++v) {
        const int kl = 2 * kb + 2 * v;
        const int pk = ks * 32 + kl;
        af.u[v] = (scls[pk] == m ? 0x3C00u : 0u) | (scls[pk + 1] == m ? 0x3C000000u : 0u);
      }
      v16h b0, b1;                                       // e^T fragments
      const int kbase = ks * 32 + g * 16;
      #pragma unroll
      for (int v = 0; v < 8; ++v) {
        float2 f0 = ((const float2*)&et[(dh * 32 + (lane & 15)) * TILE + kbase])[v];
        float2 f1 = ((const float2*)&et[(dh * 32 + 16 + (lane & 15)) * TILE + kbase])[v];
        b0[2 * v] = (_Float16)f0.x; b0[2 * v + 1] = (_Float16)f0.y;
        b1[2 * v] = (_Float16)f1.x; b1[2 * v + 1] = (_Float16)f1.y;
      }
      acc0 = __builtin_amdgcn_wmma_f32_16x16x32_f16(false, af.h, false, b0, (short)0, acc0, false, false);
      acc1 = __builtin_amdgcn_wmma_f32_16x16x32_f16(false, af.h, false, b1, (short)0, acc1, false, false);
    }
  }
  // flush: D layout -> class c = mt*16 + r + 8*g, dim = dh*32 + lane%16 (+16)
  float* wsi = ws + (size_t)n * ITEM_STRIDE;
  #pragma unroll
  for (int r = 0; r < 8; ++r) {
    const int c  = mt * 16 + r + 8 * g;
    const int d0 = dh * 32 + (lane & 15);
    atomicAdd(&wsi[OFF_SUM + c * EDIM + d0],      acc0[r]);
    atomicAdd(&wsi[OFF_SUM + c * EDIM + d0 + 16], acc1[r]);
  }
  __syncthreads();
  if (tid < CDIM) atomicAdd(&wsi[OFF_CNT + tid], lcnt[tid]);
}

// ---- pass 2: means, push/reg terms, and WMMA A-fragment packing ----
__global__ void means_kernel(float* __restrict__ ws, int n) {
  __shared__ float smean[CDIM * EDIM];
  __shared__ float sm2s[CDIM];
  __shared__ float scnt[CDIM];
  __shared__ float racc, dacc;
  const int tid = threadIdx.x;
  float* wsi = ws + (size_t)n * ITEM_STRIDE;
  if (tid < CDIM) scnt[tid] = fmaxf(wsi[OFF_CNT + tid], 1.0f);
  if (tid == 0) { racc = 0.0f; dacc = 0.0f; }
  __syncthreads();
  for (int i = tid; i < CDIM * EDIM; i += 256) {
    float m = wsi[OFF_SUM + i] / scnt[i >> 6];
    smean[i] = m;
    wsi[OFF_MEAN + i] = m;
  }
  __syncthreads();
  if (tid < CDIM) {
    float s = 0.0f;
    for (int d = 0; d < EDIM; ++d) { float m = smean[tid * EDIM + d]; s += m * m; }
    sm2s[tid] = s;
    wsi[OFF_M2 + tid] = s;
    atomicAdd(&racc, sqrtf(s + 1e-12f));
  }
  __syncthreads();
  // Pack A fragments: ks=0,1 hold -2*means; ks=2 holds [m2, 1] in K=0,1 so with
  // B2=[1, e2] the WMMA accumulator is d2 = e2 + m2 - 2*m.e directly.
  unsigned* Ap = (unsigned*)(wsi + OFF_A);
  for (int q = tid; q < 3072; q += 256) {
    const int mt    = q / 768;
    const int rem   = q % 768;
    const int ks    = rem >> 8;
    const int lane  = (rem >> 3) & 31;
    const int v     = rem & 7;
    const int m     = mt * 16 + (lane & 15);
    const int kbl   = (lane >> 4) * 8;
    const int kpair = 2 * kbl + 2 * v;
    unsigned u = 0;
    if (ks < 2) {
      const int k = ks * 32 + kpair;
      _Float16 lo = (_Float16)(-2.0f * smean[m * EDIM + k]);
      _Float16 hi = (_Float16)(-2.0f * smean[m * EDIM + k + 1]);
      u = ((unsigned)*(const unsigned short*)&hi << 16) | (unsigned)*(const unsigned short*)&lo;
    } else if (kpair == 0) {
      _Float16 lo = (_Float16)sm2s[m];
      u = 0x3C000000u | (unsigned)*(const unsigned short*)&lo;   // hi = f16(1.0)
    }
    Ap[q] = u;
  }
  // push (distance) term
  float local = 0.0f;
  for (int q = tid; q < CDIM * CDIM; q += 256) {
    const int i = q >> 6, j = q & 63;
    if (i != j) {
      float s = 0.0f;
      for (int d = 0; d < EDIM; ++d) {
        float df = smean[i * EDIM + d] - smean[j * EDIM + d];
        s += df * df;
      }
      float h = fmaxf(4.0f - sqrtf(s + 1e-12f), 0.0f);
      local += h * h;
    }
  }
  atomicAdd(&dacc, local);
  __syncthreads();
  if (tid == 0) {
    wsi[OFF_ACC + 1] = dacc / (float)(CDIM * (CDIM - 1));
    wsi[OFF_ACC + 2] = racc;
  }
}

// ---- pass 3: fused pull term + instance term; WMMA yields d2 directly ----
__global__ void main_kernel(const float* __restrict__ emb, const int* __restrict__ tgt,
                            float* __restrict__ ws, int n) {
  __shared__ __align__(16) float etile[2][EDIM * TILE];  // TDM double buffer
  __shared__ float smean[CDIM * EDIM];
  __shared__ float scnt[CDIM];
  __shared__ float se2[TILE];
  __shared__ int   scls[TILE];
  __shared__ float accI[CDIM];
  __shared__ float accD[CDIM];
  __shared__ float varacc;
  const int tid  = threadIdx.x;
  const int lane = tid & 31;
  const int g    = lane >> 4;
  const int wv   = tid >> 5;
  const int px   = (wv * 16) + (lane & 15);
  float* wsi = ws + (size_t)n * ITEM_STRIDE;
  const unsigned* Ap = (const unsigned*)(wsi + OFF_A);

  for (int i = tid; i < CDIM * EDIM; i += 256) smean[i] = wsi[OFF_MEAN + i];
  if (tid < CDIM) {
    scnt[tid] = fmaxf(wsi[OFF_CNT + tid], 1.0f);
    accI[tid] = 0.0f; accD[tid] = 0.0f;
  }
  if (tid == 0) varacc = 0.0f;
  const size_t ebase = (size_t)n * EDIM * PPIX;
  const int tile0 = blockIdx.x * ITERS;
  if (wv == 0) tdm_load_tile(emb + ebase + (size_t)tile0 * TILE, &etile[0][0]);

  for (int it = 0; it < ITERS; ++it) {
    const int p0 = (tile0 + it) * TILE;
    if (wv == 0) __builtin_amdgcn_s_wait_tensorcnt(0);
    __syncthreads();
    if (wv == 0 && it + 1 < ITERS)
      tdm_load_tile(emb + ebase + (size_t)(p0 + TILE), &etile[(it + 1) & 1][0]);
    if (tid < TILE) scls[tid] = tgt[n * PPIX + p0 + tid];
    __syncthreads();
    const float* et = &etile[it & 1][0];
    // exact f32 pull term + ||e||^2 (wave-reduced before the single LDS atomic)
    if (tid < TILE) {
      const int c = scls[tid];
      float s2 = 0.0f, sv = 0.0f;
      for (int d = 0; d < EDIM; ++d) {
        float v = et[d * TILE + tid];
        s2 += v * v;
        float df = v - smean[c * EDIM + d];
        sv += df * df;
      }
      se2[tid] = s2;
      float h = fmaxf(sqrtf(sv + 1e-12f) - 0.5f, 0.0f);
      float hv = h * h / scnt[c];
      #pragma unroll
      for (int s = 16; s >= 1; s >>= 1) hv += __shfl_xor(hv, s, 32);
      if (lane == 0) atomicAdd(&varacc, hv);
    }
    __syncthreads();
    // WMMA: acc = e2 + m2 - 2*m.e = d2.  4 independent accumulators (no RAW NOPs).
    const float e2 = se2[px];
    const int  myc = scls[px];
    v8f acc[4] = { {}, {}, {}, {} };
    #pragma unroll
    for (int ks = 0; ks < 3; ++ks) {
      v16h b;
      if (ks < 2) {
        #pragma unroll
        for (int v = 0; v < 8; ++v) {
          const int k = ks * 32 + g * 16 + 2 * v;
          b[2 * v]     = (_Float16)et[k * TILE + px];
          b[2 * v + 1] = (_Float16)et[(k + 1) * TILE + px];
        }
      } else {
        #pragma unroll
        for (int v = 0; v < 16; ++v) b[v] = (_Float16)0.0f;
        b[0] = (g == 0) ? (_Float16)1.0f : (_Float16)0.0f;   // pairs with m2 row
        b[1] = (g == 0) ? (_Float16)e2  : (_Float16)0.0f;    // pairs with ones row
      }
      #pragma unroll
      for (int mt = 0; mt < 4; ++mt) {
        union { v16h h; unsigned u[8]; } af;
        #pragma unroll
        for (int v = 0; v < 8; ++v)
          af.u[v] = Ap[((mt * 3 + ks) * 32 + lane) * 8 + v];   // L2-resident b128 loads
        acc[mt] = __builtin_amdgcn_wmma_f32_16x16x32_f16(false, af.h, false, b,
                                                         (short)0, acc[mt], false, false);
      }
    }
    // epilogue: pm = exp(-max(d2,0)/2sig); cross-lane reduce -> conflict-free ds_add
    #pragma unroll
    for (int mt = 0; mt < 4; ++mt) {
      const int cb = mt * 16 + 8 * g;
      #pragma unroll
      for (int r = 0; r < 8; ++r) {
        const int c = cb + r;
        float d2  = fmaxf(acc[mt][r], 0.0f);
        float pm  = __expf(-d2 * INV2SIG);
        float pm2 = pm * pm;
        float pmi = (myc == c) ? pm : 0.0f;
        #pragma unroll
        for (int s = 8; s >= 1; s >>= 1) {
          pm2 += __shfl_xor(pm2, s, 32);
          pmi += __shfl_xor(pmi, s, 32);
        }
        if ((lane & 15) == 0) {
          atomicAdd(&accD[c], pm2);
          atomicAdd(&accI[c], pmi);
        }
      }
    }
  }
  __syncthreads();
  if (tid < CDIM) {
    atomicAdd(&wsi[OFF_ACC + 3 + tid],  accI[tid]);
    atomicAdd(&wsi[OFF_ACC + 67 + tid], accD[tid]);
  }
  if (tid == 0) atomicAdd(&wsi[OFF_ACC + 0], varacc);
}

// ---- pass 4: combine terms -> scalar loss ----
__global__ void finalize_kernel(const float* __restrict__ ws, float* __restrict__ out) {
  __shared__ float dsum;
  const int tid = threadIdx.x;    // 64 threads, one per class
  float total = 0.0f;
  for (int n = 0; n < NITEMS; ++n) {
    const float* wsi = ws + (size_t)n * ITEM_STRIDE;
    if (tid == 0) dsum = 0.0f;
    __syncthreads();
    float inter = wsi[OFF_ACC + 3 + tid];
    float den   = wsi[OFF_ACC + 67 + tid] + wsi[OFF_CNT + tid];
    atomicAdd(&dsum, 2.0f * inter / fmaxf(den, 1e-6f));
    __syncthreads();
    if (tid == 0) {
      float var_term  = wsi[OFF_ACC + 0] / (float)CDIM;
      float dist_term = wsi[OFF_ACC + 1];
      float reg_term  = 0.001f * wsi[OFF_ACC + 2] / (float)CDIM;
      float inst_term = 1.0f - dsum / (float)CDIM;
      total += var_term + dist_term + reg_term + inst_term;
    }
    __syncthreads();
  }
  if (tid == 0) out[0] = total / (float)NITEMS;
}

extern "C" void kernel_launch(void* const* d_in, const int* in_sizes, int n_in,
                              void* d_out, int out_size, void* d_ws, size_t ws_size,
                              hipStream_t stream) {
  const float* emb = (const float*)d_in[0];
  const int*   tgt = (const int*)d_in[1];
  float* ws  = (float*)d_ws;     // 4 * 11584 * 4 B ~= 185 KB
  float* out = (float*)d_out;
  zero_ws_kernel<<<128, 256, 0, stream>>>(ws, NITEMS * ITEM_STRIDE);
  // per-item sequencing: one item's embeddings (64 MB) stay resident in the
  // 192 MB L2 between the stats pass and the main pass -> ~halved HBM traffic.
  for (int n = 0; n < NITEMS; ++n) {
    stats_kernel<<<NBLK, 256, 0, stream>>>(emb, tgt, ws, n);
    means_kernel<<<1, 256, 0, stream>>>(ws, n);
    main_kernel <<<NBLK, 256, 0, stream>>>(emb, tgt, ws, n);
  }
  finalize_kernel<<<1, 64, 0, stream>>>(ws, out);
}
